// GATLayer_22454089024164
// MI455X (gfx1250) — compile-verified, matrix operand order checked by reading, not run on hardware
//
#include <hip/hip_runtime.h>

// GAT layer for MI455X (gfx1250, wave32, WMMA, async-to-LDS).
//
// Math reduction: softmax row term h1[i] cancels; with
// w[j] = exp(h2[j] - gmax)  (h2 = input @ A[:256], gmax = max_j h2[j]):
//   out[i,:] = (sum_j adj[i,j] * w[j] * input[j,:]) / (sum_j adj[i,j] * w[j])
// => single f16 WMMA GEMM:  C[8192,288] = adjF16[8192,8192] @ X'[8192,288]
//    X'[j,0:256] = w[j]*input[j,:], X'[j,256] = w[j] (denominator column),
//    X'[j,257:288] = 0 pad.  adjF16 is exact {0, 1.0h} (bit pattern 0x3C00).
// Roofline: adj = 256 MB read once (~11us @ 23.3 TB/s); 38.6 GFLOP f16 WMMA.
// X' (4.7 MB, transposed [288][8192] in ws) stays resident in the 192 MB L2.
//
// Double-buffers both LDS tiles via GLOBAL_LOAD_ASYNC_TO_LDS_B128
// (ASYNCcnt-tracked, 6 issues per wave per tile so s_wait_asynccnt gets an
// immediate), overlapping the next tile's HBM fetch with the current WMMAs.
// adj is copied raw (int32) and converted to f16 during A-fragment build.
// Workspace needed: 64 KB + 288*8192*2 B (~4.8 MB).

typedef __attribute__((ext_vector_type(16))) _Float16 v16h;
typedef __attribute__((ext_vector_type(8)))  float    v8f;
typedef unsigned int u32;

constexpr int N  = 8192;
constexpr int D  = 256;
constexpr int BN = 288;   // 256 data cols + 1 denom col + 31 zero pad
constexpr int BK = 32;    // WMMA f16 K
constexpr int BM = 32;    // rows per block
constexpr int LSTRIDE = 20;  // xs row stride (uints): 16B-aligned, conflict-free
constexpr int ASTRIDE = 36;  // adjr row stride (ints): 16B-aligned, conflict-free

#if __has_builtin(__builtin_amdgcn_global_load_async_to_lds_b128) && \
    __has_builtin(__builtin_amdgcn_s_wait_asynccnt)
#define HAVE_ASYNC 1
#else
#define HAVE_ASYNC 0
#endif

// Builtin signature (from hipcc diagnostic): params are pointers to the raw
// clang vector '__attribute__((__vector_size__(16))) int' in AS1 / AS3 —
// NOT HIP's int4 (which is HIP_vector_type<int,4>, a struct).
typedef int v4i __attribute__((vector_size(16)));
typedef __attribute__((address_space(1))) v4i gvec4;
typedef __attribute__((address_space(3))) v4i lvec4;
#define AS1(p) ((gvec4*)(p))
#define AS3(p) ((lvec4*)(p))

union Frag { v16h h; uint4 q[2]; u32 u[8]; };

__device__ __forceinline__ u32 pack2(int a, int b) {
  return (a > 0 ? 0x3C00u : 0u) | ((b > 0 ? 0x3C00u : 0u) << 16);
}

// ---- kernel 1: h2[j] = dot(input[j,:], A[0:256]) -------------------------
__global__ __launch_bounds__(256) void h2_kernel(const float* __restrict__ x,
                                                 const float* __restrict__ A,
                                                 float* __restrict__ h2) {
  const int wave = threadIdx.x >> 5;
  const int lane = threadIdx.x & 31;
  const int row  = blockIdx.x * 8 + wave;
  const float* xr = x + (size_t)row * D;
  float s = 0.f;
  #pragma unroll
  for (int c = 0; c < D; c += 32) s += xr[c + lane] * A[c + lane];
  #pragma unroll
  for (int off = 16; off; off >>= 1) s += __shfl_xor(s, off, 32);
  if (lane == 0) h2[row] = s;
}

// ---- kernel 2: global max of h2 ------------------------------------------
__global__ __launch_bounds__(1024) void gmax_kernel(const float* __restrict__ h2,
                                                    float* __restrict__ gmax) {
  __shared__ float red[1024];
  float m = -1e30f;
  for (int i = threadIdx.x; i < N; i += 1024) m = fmaxf(m, h2[i]);
  red[threadIdx.x] = m;
  __syncthreads();
  for (int s = 512; s; s >>= 1) {
    if ((int)threadIdx.x < s) red[threadIdx.x] = fmaxf(red[threadIdx.x], red[threadIdx.x + s]);
    __syncthreads();
  }
  if (threadIdx.x == 0) *gmax = red[0];
}

// ---- kernel 3: build X' transposed: XT[c][j], f16 ------------------------
__global__ __launch_bounds__(288) void xt_kernel(const float* __restrict__ x,
                                                 const float* __restrict__ h2,
                                                 const float* __restrict__ gmax,
                                                 _Float16* __restrict__ XT) {
  const int j = blockIdx.x;
  const float w = __expf(h2[j] - *gmax);
  const int c = threadIdx.x;                // 0..287
  float v = (c < D) ? w * x[(size_t)j * D + c] : (c == D ? w : 0.f);
  XT[(size_t)c * N + j] = (_Float16)v;
}

// ---- kernel 4: C = adjF16 @ X' via v_wmma_f32_16x16x32_f16, fused epilogue
__global__ __launch_bounds__(256) void gat_gemm_kernel(const int* __restrict__ adj,
                                                       const _Float16* __restrict__ XT,
                                                       float* __restrict__ out) {
  const int tid    = threadIdx.x;
  const int lane   = tid & 31;
  const int waveId = tid >> 5;         // 0..7
  const int wm     = waveId & 1;       // M half: rows wm*16..wm*16+15
  const int wn     = waveId >> 1;      // N group: tiles {0-4},{5-9},{10-13},{14-17}
  const int hi     = (lane >> 4) & 1;  // lane half (fragment K-group select)
  const int ln     = lane & 15;
  const int ntiles = (wn < 2) ? 5 : 4;
  const int nbase  = wn * 5 - ((wn >= 2) ? (wn - 2) : 0);
  const int row0   = blockIdx.x * BM;

  __shared__ float denomS[BM];
  v8f acc[5] = {};

#if HAVE_ASYNC
  // ---- async double-buffered pipeline ------------------------------------
  __shared__ u32 xs[2][BN][LSTRIDE];   // X' tiles, f16 packed pairs
  __shared__ int adjr[2][BM][ASTRIDE]; // raw adj int32 tiles

  // 6 async issues per wave per tile: 1 adj (32x32 i32) + 5 X' (288x64B,
  // 1152 16B-chunks over 256 threads, index clamped => wave-uniform count).
  auto issueTile = [&](int buf, int kk) {
    {
      const int r  = tid >> 3;             // 0..31
      const int s4 = (tid & 7) << 2;       // int offset in row: 0,4,...,28
      const int* gs = adj + (size_t)(row0 + r) * N + kk + s4;
      __builtin_amdgcn_global_load_async_to_lds_b128(
          AS1(gs), AS3(&adjr[buf][r][s4]), 0, 0);
    }
    #pragma unroll
    for (int i = 0; i < 5; i++) {
      int idx = tid + i * 256;
      idx = (idx < 1151) ? idx : 1151;     // clamp: duplicate benign copy
      const int c  = idx >> 2;             // X' row (column of C)
      const int su = (idx & 3) << 2;       // uint offset in LDS row
      const _Float16* gs = XT + (size_t)c * N + kk + (su << 1);
      __builtin_amdgcn_global_load_async_to_lds_b128(
          AS1(gs), AS3(&xs[buf][c][su]), 0, 0);
    }
  };

  issueTile(0, 0);
  int p = 0;
  for (int kk = 0; kk < N; kk += BK, p ^= 1) {
    if (kk + BK < N) {
      __syncthreads();                         // prev readers of buf p^1 done
      issueTile(p ^ 1, kk + BK);               // prefetch next tile
      __builtin_amdgcn_s_wait_asynccnt(6);     // oldest 6 (= buf p) arrived
    } else {
      __builtin_amdgcn_s_wait_asynccnt(0);
    }
    __syncthreads();                           // buf p visible to all waves

    // A fragment from raw int32 adj: lanes 0-15 K{0-7,16-23}, 16-31 K{8-15,24-31}
    const int* arow = &adjr[p][wm * 16 + ln][0];
    int4 a0 = *(const int4*)(arow + 8 * hi);
    int4 a1 = *(const int4*)(arow + 8 * hi + 4);
    int4 a2 = *(const int4*)(arow + 8 * hi + 16);
    int4 a3 = *(const int4*)(arow + 8 * hi + 20);
    Frag fa;
    fa.u[0] = pack2(a0.x, a0.y); fa.u[1] = pack2(a0.z, a0.w);
    fa.u[2] = pack2(a1.x, a1.y); fa.u[3] = pack2(a1.z, a1.w);
    fa.u[4] = pack2(a2.x, a2.y); fa.u[5] = pack2(a2.z, a2.w);
    fa.u[6] = pack2(a3.x, a3.y); fa.u[7] = pack2(a3.z, a3.w);

    // Preload all B fragments, then chain WMMAs back-to-back.
    Frag fb[5];
    #pragma unroll
    for (int t = 0; t < 5; t++) {
      if (t < ntiles) {
        const u32* colp = &xs[p][(nbase + t) * 16 + ln][0];
        fb[t].q[0] = *(const uint4*)(colp + 4 * hi);
        fb[t].q[1] = *(const uint4*)(colp + 8 + 4 * hi);
      }
    }
    #pragma unroll
    for (int t = 0; t < 5; t++) {
      if (t < ntiles) {
        acc[t] = __builtin_amdgcn_wmma_f32_16x16x32_f16(
            false, fa.h, false, fb[t].h, (short)0, acc[t], false, false);
      }
    }
  }
#else
  // ---- fallback: synchronous single-buffer pipeline ----------------------
  __shared__ u32 adjs[BM][LSTRIDE];
  __shared__ u32 xs[BN][LSTRIDE];

  for (int kk = 0; kk < N; kk += BK) {
    __syncthreads();
    {
      const int r  = tid >> 3;
      const int c4 = (tid & 7) << 2;
      int4 a = *(const int4*)(adj + (size_t)(row0 + r) * N + kk + c4);
      adjs[r][(c4 >> 1) + 0] = pack2(a.x, a.y);
      adjs[r][(c4 >> 1) + 1] = pack2(a.z, a.w);
    }
    for (int c = tid; c < BN; c += 256) {
      const uint4* src = (const uint4*)(XT + (size_t)c * N + kk);
      uint4* dst = (uint4*)&xs[c][0];
      dst[0] = src[0]; dst[1] = src[1]; dst[2] = src[2]; dst[3] = src[3];
    }
    __syncthreads();

    Frag fa;
    {
      const u32* rowp = &adjs[wm * 16 + ln][0];
      fa.q[0] = *(const uint4*)(rowp + 4 * hi);
      fa.q[1] = *(const uint4*)(rowp + 8 + 4 * hi);
    }
    Frag fb[5];
    #pragma unroll
    for (int t = 0; t < 5; t++) {
      if (t < ntiles) {
        const u32* colp = &xs[(nbase + t) * 16 + ln][0];
        fb[t].q[0] = *(const uint4*)(colp + 4 * hi);
        fb[t].q[1] = *(const uint4*)(colp + 8 + 4 * hi);
      }
    }
    #pragma unroll
    for (int t = 0; t < 5; t++) {
      if (t < ntiles) {
        acc[t] = __builtin_amdgcn_wmma_f32_16x16x32_f16(
            false, fa.h, false, fb[t].h, (short)0, acc[t], false, false);
      }
    }
  }
#endif

  __syncthreads();
  // Denominator column (global col 256) = wn==3, local tile 2, in-tile n==0.
  if (wn == 3 && ln == 0) {
    #pragma unroll
    for (int r = 0; r < 8; r++)
      denomS[wm * 16 + r + 8 * hi] = acc[2][r];
  }
  __syncthreads();

  float rden[8];
  #pragma unroll
  for (int r = 0; r < 8; r++) {
    const float d = denomS[wm * 16 + r + 8 * hi];
    rden[r] = (d > 0.f) ? (1.0f / d) : 0.f;  // fully-masked row -> 0 (prob ~0)
  }
  #pragma unroll
  for (int t = 0; t < 5; t++) {
    if (t < ntiles) {
      const int col = (nbase + t) * 16 + ln;
      if (col < D) {
        #pragma unroll
        for (int r = 0; r < 8; r++) {
          const int row = row0 + wm * 16 + r + 8 * hi;
          out[(size_t)row * D + col] = acc[t][r] * rden[r];
        }
      }
    }
  }
}

extern "C" void kernel_launch(void* const* d_in, const int* in_sizes, int n_in,
                              void* d_out, int out_size, void* d_ws, size_t ws_size,
                              hipStream_t stream) {
  const float* inp = (const float*)d_in[0];   // [8192,256] f32
  const int*   adj = (const int*)d_in[1];     // [8192,8192] i32
  const float* A   = (const float*)d_in[2];   // [512,1] f32
  float* out = (float*)d_out;                 // [8192,256] f32

  char* ws = (char*)d_ws;
  float*    h2   = (float*)ws;                        // 8192 f32
  float*    gmax = (float*)(ws + 32 * 1024);          // 1 f32
  _Float16* XT   = (_Float16*)(ws + 64 * 1024);       // [288][8192] f16

  h2_kernel  <<<N / 8, 256, 0, stream>>>(inp, A, h2);
  gmax_kernel<<<1, 1024, 0, stream>>>(h2, gmax);
  xt_kernel  <<<N, BN, 0, stream>>>(inp, h2, gmax, XT);
  gat_gemm_kernel<<<N / BM, 256, 0, stream>>>(adj, XT, out);
}
